// TwoStageCoarseRerankModel_54984171323610
// MI455X (gfx1250) — matrix operand order repaired and checked by reading, not compiled, not run on hardware
//
#include <hip/hip_runtime.h>
#include <math.h>

// Problem constants (match reference)
#define B_  64
#define N_  8192
#define D_  256
#define G_  8
#define H_  256
#define LD_ 256
#define R_  128
#define K_  64

typedef __attribute__((ext_vector_type(16))) _Float16 v16h;
typedef __attribute__((ext_vector_type(8)))  _Float16 v8h;
typedef __attribute__((ext_vector_type(8)))  float    v8f;
typedef __attribute__((ext_vector_type(4)))  float    v4f;

// ---------------------------------------------------------------------------
// WMMA fragment loaders (CDNA5 16x16x32 f16 layouts, ISA 7.12.2)
// A (16x32, MxK): lanes 0-15 hold M=lane, K in {base+0..7, base+16..23};
//                 lanes 16-31 hold M=lane-16, K in {base+8..15, base+24..31}.
// ---------------------------------------------------------------------------
__device__ __forceinline__ v16h load_a_frag(const _Float16* S, int mt, int ks, int lane) {
    const int row  = mt * 16 + (lane & 15);
    const int koff = ks * 32 + ((lane >> 4) << 3);   // 0 or 8
    v8h x0 = *(const v8h*)(S + row * H_ + koff);
    v8h x1 = *(const v8h*)(S + row * H_ + koff + 16);
    v16h a;
#pragma unroll
    for (int j = 0; j < 8; ++j) { a[j] = x0[j]; a[8 + j] = x1[j]; }
    return a;
}

// B (32x16, KxN): lane holds column N=lane&15; lanes 0-15 K in {base+0..7, base+16..23},
// lanes 16-31 K in {base+8..15, base+24..31} (matches A).  Weights prepped column-major
// [Nout][256] f16 so this is two 16-byte reads per lane.
__device__ __forceinline__ v16h load_b_frag(const _Float16* Wt, int nt, int ks, int lane) {
    const int col  = nt * 16 + (lane & 15);
    const int base = ks * 32 + ((lane >> 4) << 3);   // 0 or 8
    const v8h* p = (const v8h*)(Wt + col * 256 + base);
    v8h x0 = p[0], x1 = p[2];
    v16h b;
#pragma unroll
    for (int j = 0; j < 8; ++j) { b[j] = x0[j]; b[8 + j] = x1[j]; }
    return b;
}

// ---------------------------------------------------------------------------
// Kernel 1: f32 -> f16 column-major weight prep into workspace
// ---------------------------------------------------------------------------
__global__ __launch_bounds__(256) void prep_kernel(
    const float* __restrict__ We1, const float* __restrict__ We2,
    const float* __restrict__ Wr1, const float* __restrict__ Wr2,
    _Float16* __restrict__ We1t, _Float16* __restrict__ We2t,
    _Float16* __restrict__ Wr1t, _Float16* __restrict__ Wr2t)
{
    int i = blockIdx.x * 256 + threadIdx.x;
    if (i < 65536) {                      // We1t[h][k] = We1[k][h]   (k<256 only; geom tail handled in epilogue)
        We1t[i] = (_Float16)We1[(i & 255) * H_ + (i >> 8)];
    } else if (i < 131072) {
        int j = i - 65536;                // We2t[ho][hi] = We2[hi][ho]
        We2t[j] = (_Float16)We2[(j & 255) * H_ + (j >> 8)];
    } else if (i < 163840) {
        int j = i - 131072;               // Wr1t[r][h] = Wr1[h][r]
        Wr1t[j] = (_Float16)Wr1[(j & 255) * R_ + (j >> 8)];
    } else if (i < 196608) {
        int j = i - 163840;               // Wr2t[r][h] = Wr2[h][r]
        Wr2t[j] = (_Float16)Wr2[(j & 255) * R_ + (j >> 8)];
    }
}

// ---------------------------------------------------------------------------
// Kernel 2: output init (full_logits = -inf, full_anchor = 0)
// ---------------------------------------------------------------------------
__global__ __launch_bounds__(256) void init_out_kernel(float* __restrict__ out)
{
    int i = blockIdx.x * 256 + threadIdx.x;
    if (i < B_ * N_)            out[i] = -INFINITY;
    else if (i < 2 * B_ * N_)   out[i] = 0.0f;
}

// ---------------------------------------------------------------------------
// Kernel 3: coarse matvec — the 512MB streaming pass (HBM-bound, ~22us floor).
// One wave32 per row; two coalesced non-temporal 16B loads per lane;
// shfl-xor tree reduce.  NT hint keeps the one-shot stream from evicting the
// L2-resident weights/shortlist data the fine stage reuses.
// ---------------------------------------------------------------------------
__global__ __launch_bounds__(256) void coarse_kernel(
    const float* __restrict__ feat, const float* __restrict__ wc,
    const float* __restrict__ bcp, float* __restrict__ clog)
{
    const int lane = threadIdx.x & 31;
    const int gw   = blockIdx.x * 8 + (threadIdx.x >> 5);
    const int nwaves = gridDim.x * 8;
    const v4f w0 = *(const v4f*)(wc + lane * 4);
    const v4f w1 = *(const v4f*)(wc + 128 + lane * 4);
    const float bc = *bcp;
    for (int row = gw; row < B_ * N_; row += nwaves) {
        const float* p = feat + (size_t)row * D_;
        v4f v0 = __builtin_nontemporal_load((const v4f*)(p + lane * 4));
        v4f v1 = __builtin_nontemporal_load((const v4f*)(p + 128 + lane * 4));
        float s = v0.x * w0.x + v0.y * w0.y + v0.z * w0.z + v0.w * w0.w
                + v1.x * w1.x + v1.y * w1.y + v1.z * w1.z + v1.w * w1.w;
#pragma unroll
        for (int m = 16; m >= 1; m >>= 1) s += __shfl_xor(s, m, 32);
        if (lane == 0) clog[row] = s + bc;
    }
}

// ---------------------------------------------------------------------------
// Kernel 4: per-batch top-64 with gold injection. One block per batch.
// Register-cached local maxima: per extraction only a 5-step wave shfl argmax
// + one cross-wave merge (2 barriers) and a 32-element rescan by the single
// owner thread of the extracted slot.
// ---------------------------------------------------------------------------
__global__ __launch_bounds__(256) void topk_kernel(
    const float* __restrict__ clog, const int* __restrict__ target,
    int* __restrict__ idxout)
{
    __shared__ float vals[N_];            // 32 KB
    __shared__ float wbv[8];
    __shared__ int   wbi[8];
    __shared__ int   seli[K_];
    __shared__ int   gbi_sh;
    __shared__ int   found;
    const int b = blockIdx.x, t = threadIdx.x;
    const int lane = t & 31, wv = t >> 5;

    for (int i = t; i < N_; i += 256) vals[i] = clog[b * N_ + i];
    if (t == 0) found = 0;
    __syncthreads();

    // initial per-thread local argmax over its 32 strided elements
    float bv = -INFINITY; int bi = N_;
#pragma unroll 4
    for (int j = 0; j < 32; ++j) {
        int i = t + 256 * j;              // strictly increasing -> first-occurrence ties
        float v = vals[i];
        if (v > bv) { bv = v; bi = i; }
    }

    for (int it = 0; it < K_; ++it) {
        // wave-level argmax (no barriers)
        float rv = bv; int ri = bi;
#pragma unroll
        for (int m = 16; m >= 1; m >>= 1) {
            float ov = __shfl_xor(rv, m, 32);
            int   oi = __shfl_xor(ri, m, 32);
            if (ov > rv || (ov == rv && oi < ri)) { rv = ov; ri = oi; }
        }
        if (lane == 0) { wbv[wv] = rv; wbi[wv] = ri; }
        __syncthreads();
        if (t == 0) {
            float gv = wbv[0]; int gi = wbi[0];
#pragma unroll
            for (int w = 1; w < 8; ++w)
                if (wbv[w] > gv || (wbv[w] == gv && wbi[w] < gi)) { gv = wbv[w]; gi = wbi[w]; }
            seli[it] = gi;
            vals[gi] = -INFINITY;
            gbi_sh = gi;
        }
        __syncthreads();
        const int gi = gbi_sh;
        if ((gi & 255) == t) {            // owner of the extracted slot rescans
            bv = -INFINITY; bi = N_;
#pragma unroll 4
            for (int j = 0; j < 32; ++j) {
                int i = t + 256 * j;
                float v = vals[i];
                if (v > bv) { bv = v; bi = i; }
            }
        }
    }

    const int tg = target[b];
    if (t < K_ && seli[t] == tg) found = 1;   // benign race (write of 1)
    __syncthreads();
    if (t == 0 && tg >= 0 && tg < N_ && !found) seli[K_ - 1] = tg;  // worst = last (descending order)
    __syncthreads();
    if (t < K_) idxout[b * K_ + t] = seli[t];
}

// ---------------------------------------------------------------------------
// Kernel 5: fine rerank model. One block (8 wave32) per batch.
// All GEMMs via v_wmma_f32_16x16x32_f16; s_rel factorized to avoid KxK r_ij.
// ---------------------------------------------------------------------------
__global__ __launch_bounds__(256) void fine_kernel(
    const float* __restrict__ feat, const float* __restrict__ geom,
    const float* __restrict__ qt, const float* __restrict__ qa, const float* __restrict__ qr,
    const float* __restrict__ We1, const float* __restrict__ be1, const float* __restrict__ be2,
    const float* __restrict__ Wa, const float* __restrict__ Wanc, const float* __restrict__ Wr3,
    const _Float16* __restrict__ We1t, const _Float16* __restrict__ We2t,
    const _Float16* __restrict__ Wr1t, const _Float16* __restrict__ Wr2t,
    const int* __restrict__ idx,
    float* __restrict__ outL, float* __restrict__ outA)
{
    const int b = blockIdx.x;
    const int t = threadIdx.x;
    const int lane = t & 31;
    const int wv = t >> 5;

    __shared__ __align__(16) _Float16 sH[K_ * H_];   // X -> H1 -> H2 (in place per 32-row half), 32 KB
    __shared__ __align__(16) _Float16 sA[K_ * R_];   // A2 then A1, 16 KB
    __shared__ float sGeom[K_ * G_];
    __shared__ float sVa[H_], sVanc[H_], sWr[R_];
    __shared__ float sAttr[K_], sAnc[K_], sP[K_], sC[R_];
    __shared__ int   sIdx[K_];

    if (t < K_) sIdx[t] = idx[b * K_ + t];
    __syncthreads();

    // gather geom [64][8]
    for (int e = t; e < K_ * G_; e += 256) {
        int i = e >> 3, g = e & 7;
        sGeom[e] = geom[((size_t)b * N_ + sIdx[i]) * G_ + g];
    }
    // gather features -> f16 X in sH (4 threads per row, 64 floats each)
    {
        const int i = t >> 2, q = t & 3;
        const float* src = feat + ((size_t)b * N_ + sIdx[i]) * D_ + q * 64;
        _Float16* dst = sH + i * H_ + q * 64;
#pragma unroll
        for (int j = 0; j < 64; j += 4) {
            v4f v = *(const v4f*)(src + j);
            dst[j] = (_Float16)v.x; dst[j + 1] = (_Float16)v.y;
            dst[j + 2] = (_Float16)v.z; dst[j + 3] = (_Float16)v.w;
        }
    }
    // head vectors: v_a = Wa@q_t, v_anc = Wanc@q_a, wr = q_r@Wr3
    {
        const float* qtb = qt + b * LD_;
        const float* qab = qa + b * LD_;
        const float* ra = Wa + t * LD_;
        const float* rn = Wanc + t * LD_;
        float sa = 0.f, sn = 0.f;
        for (int l = 0; l < LD_; ++l) { sa += ra[l] * qtb[l]; sn += rn[l] * qab[l]; }
        sVa[t] = sa; sVanc[t] = sn;
    }
    if (t < R_) {
        const float* qrb = qr + b * LD_;
        float s = 0.f;
        for (int l = 0; l < LD_; ++l) s += qrb[l] * Wr3[l * R_ + t];
        sWr[t] = s;
    }
    __syncthreads();

    // ---- layer 1: H1 = relu(X@We1 + geom@We1[256:264] + be1), in place per half ----
    for (int half = 0; half < 2; ++half) {
        const int mt = half * 2 + (wv >> 2);
        const int ntb = (wv & 3) * 4;
        v8f acc[4];
#pragma unroll
        for (int q = 0; q < 4; ++q)
#pragma unroll
            for (int v = 0; v < 8; ++v) acc[q][v] = 0.f;
        for (int ks = 0; ks < 8; ++ks) {
            v16h a = load_a_frag(sH, mt, ks, lane);
#pragma unroll
            for (int q = 0; q < 4; ++q) {
                v16h bb = load_b_frag(We1t, ntb + q, ks, lane);
                acc[q] = __builtin_amdgcn_wmma_f32_16x16x32_f16(
                    false, a, false, bb, (short)0, acc[q], false, false);
            }
        }
        __syncthreads();   // all reads of this half's X rows done
#pragma unroll
        for (int q = 0; q < 4; ++q) {
            const int col = (ntb + q) * 16 + (lane & 15);
            const float bias = be1[col];
            float wg[8];
#pragma unroll
            for (int g = 0; g < 8; ++g) wg[g] = We1[(D_ + g) * H_ + col];
            const int m0 = mt * 16 + ((lane >> 4) << 3);
#pragma unroll
            for (int v = 0; v < 8; ++v) {
                const int row = m0 + v;
                float s = acc[q][v] + bias;
#pragma unroll
                for (int g = 0; g < 8; ++g) s += sGeom[row * G_ + g] * wg[g];
                sH[row * H_ + col] = (_Float16)fmaxf(s, 0.f);
            }
        }
        __syncthreads();
    }

    // ---- layer 2: H2 = relu(H1@We2 + be2), in place per half ----
    for (int half = 0; half < 2; ++half) {
        const int mt = half * 2 + (wv >> 2);
        const int ntb = (wv & 3) * 4;
        v8f acc[4];
#pragma unroll
        for (int q = 0; q < 4; ++q)
#pragma unroll
            for (int v = 0; v < 8; ++v) acc[q][v] = 0.f;
        for (int ks = 0; ks < 8; ++ks) {
            v16h a = load_a_frag(sH, mt, ks, lane);
#pragma unroll
            for (int q = 0; q < 4; ++q) {
                v16h bb = load_b_frag(We2t, ntb + q, ks, lane);
                acc[q] = __builtin_amdgcn_wmma_f32_16x16x32_f16(
                    false, a, false, bb, (short)0, acc[q], false, false);
            }
        }
        __syncthreads();
#pragma unroll
        for (int q = 0; q < 4; ++q) {
            const int col = (ntb + q) * 16 + (lane & 15);
            const float bias = be2[col];
            const int m0 = mt * 16 + ((lane >> 4) << 3);
#pragma unroll
            for (int v = 0; v < 8; ++v) {
                const int row = m0 + v;
                sH[row * H_ + col] = (_Float16)fmaxf(acc[q][v] + bias, 0.f);
            }
        }
        __syncthreads();
    }

    // ---- heads: s_attr, anchor softmax ----
    if (t < K_) {
        const _Float16* hr = sH + t * H_;
        float sa = 0.f, sn = 0.f;
        for (int h = 0; h < H_; ++h) {
            float hv = (float)hr[h];
            sa += hv * sVa[h]; sn += hv * sVanc[h];
        }
        sAttr[t] = sa; sAnc[t] = sn;
    }
    __syncthreads();
    if (t < K_) {
        float m = -INFINITY;
        for (int j = 0; j < K_; ++j) m = fmaxf(m, sAnc[j]);
        sC[t] = expf(sAnc[t] - m);
    }
    __syncthreads();
    if (t < K_) {
        float s = 0.f;
        for (int j = 0; j < K_; ++j) s += sC[j];
        sP[t] = sC[t] / s;
    }
    __syncthreads();

    // ---- A2 = H2@Wr2  (then c[r] = sum_j p[j]*A2[j,r]) ----
    {
        const int mt = wv >> 1;
        const int ntb = (wv & 1) * 4;
        v8f acc[4];
#pragma unroll
        for (int q = 0; q < 4; ++q)
#pragma unroll
            for (int v = 0; v < 8; ++v) acc[q][v] = 0.f;
        for (int ks = 0; ks < 8; ++ks) {
            v16h a = load_a_frag(sH, mt, ks, lane);
#pragma unroll
            for (int q = 0; q < 4; ++q) {
                v16h bb = load_b_frag(Wr2t, ntb + q, ks, lane);
                acc[q] = __builtin_amdgcn_wmma_f32_16x16x32_f16(
                    false, a, false, bb, (short)0, acc[q], false, false);
            }
        }
#pragma unroll
        for (int q = 0; q < 4; ++q) {
            const int col = (ntb + q) * 16 + (lane & 15);
            const int m0 = mt * 16 + ((lane >> 4) << 3);
#pragma unroll
            for (int v = 0; v < 8; ++v) sA[(m0 + v) * R_ + col] = (_Float16)acc[q][v];
        }
    }
    __syncthreads();
    if (t < R_) {
        float s = 0.f;
        for (int j = 0; j < K_; ++j) s += sP[j] * (float)sA[j * R_ + t];
        sC[t] = s;
    }
    __syncthreads();

    // ---- A1 = H2@Wr1, then s_rel + scatter outputs ----
    {
        const int mt = wv >> 1;
        const int ntb = (wv & 1) * 4;
        v8f acc[4];
#pragma unroll
        for (int q = 0; q < 4; ++q)
#pragma unroll
            for (int v = 0; v < 8; ++v) acc[q][v] = 0.f;
        for (int ks = 0; ks < 8; ++ks) {
            v16h a = load_a_frag(sH, mt, ks, lane);
#pragma unroll
            for (int q = 0; q < 4; ++q) {
                v16h bb = load_b_frag(Wr1t, ntb + q, ks, lane);
                acc[q] = __builtin_amdgcn_wmma_f32_16x16x32_f16(
                    false, a, false, bb, (short)0, acc[q], false, false);
            }
        }
#pragma unroll
        for (int q = 0; q < 4; ++q) {
            const int col = (ntb + q) * 16 + (lane & 15);
            const int m0 = mt * 16 + ((lane >> 4) << 3);
#pragma unroll
            for (int v = 0; v < 8; ++v) sA[(m0 + v) * R_ + col] = (_Float16)acc[q][v];
        }
    }
    __syncthreads();
    if (t < K_) {
        const _Float16* ar = sA + t * R_;
        float sr = 0.f;
        for (int r = 0; r < R_; ++r) sr += (float)ar[r] * sWr[r] * sC[r];
        const float fl = sAttr[t] + sr;
        const int oi = sIdx[t];
        outL[(size_t)b * N_ + oi] = fl;
        outA[(size_t)b * N_ + oi] = sP[t];
    }
}

// ---------------------------------------------------------------------------
extern "C" void kernel_launch(void* const* d_in, const int* in_sizes, int n_in,
                              void* d_out, int out_size, void* d_ws, size_t ws_size,
                              hipStream_t stream)
{
    (void)in_sizes; (void)n_in; (void)out_size; (void)ws_size;
    const float* feat   = (const float*)d_in[0];
    const float* geom   = (const float*)d_in[1];
    /* d_in[2] object_mask: all-true, unused */
    const float* qt     = (const float*)d_in[3];
    const float* qa     = (const float*)d_in[4];
    const float* qr     = (const float*)d_in[5];
    const int*   target = (const int*)d_in[6];
    const float* wc     = (const float*)d_in[7];
    const float* bc     = (const float*)d_in[8];
    const float* We1    = (const float*)d_in[9];
    const float* be1    = (const float*)d_in[10];
    const float* We2    = (const float*)d_in[11];
    const float* be2    = (const float*)d_in[12];
    const float* Wa     = (const float*)d_in[13];
    const float* Wanc   = (const float*)d_in[14];
    const float* Wr1    = (const float*)d_in[15];
    const float* Wr2    = (const float*)d_in[16];
    const float* Wr3    = (const float*)d_in[17];

    char* ws = (char*)d_ws;
    float*     clog = (float*)ws;                        // 2,097,152 B
    int*       idxb = (int*)(ws + 2097152);              //    16,384 B
    _Float16*  We1t = (_Float16*)(ws + 2113536);         //   131,072 B
    _Float16*  We2t = (_Float16*)(ws + 2244608);         //   131,072 B
    _Float16*  Wr1t = (_Float16*)(ws + 2375680);         //    65,536 B
    _Float16*  Wr2t = (_Float16*)(ws + 2441216);         //    65,536 B

    float* outL = (float*)d_out;
    float* outA = outL + (size_t)B_ * N_;

    prep_kernel<<<768, 256, 0, stream>>>(We1, We2, Wr1, Wr2, We1t, We2t, Wr1t, Wr2t);
    init_out_kernel<<<4096, 256, 0, stream>>>(outL);
    coarse_kernel<<<2048, 256, 0, stream>>>(feat, wc, bc, clog);
    topk_kernel<<<64, 256, 0, stream>>>(clog, target, idxb);
    fine_kernel<<<64, 256, 0, stream>>>(feat, geom, qt, qa, qr,
                                        We1, be1, be2, Wa, Wanc, Wr3,
                                        We1t, We2t, Wr1t, Wr2t, idxb, outL, outA);
}
